// MOGBlock_74620761801379
// MI455X (gfx1250) — matrix operand order repaired and checked by reading, not compiled, not run on hardware
//
#include <hip/hip_runtime.h>
#include <hip/hip_bf16.h>
#include <cstdint>
#include <cstddef>

// ---------------------------------------------------------------------------
// Types for gfx1250 WMMA (wave32)
// ---------------------------------------------------------------------------
typedef __attribute__((ext_vector_type(16))) __bf16 bf16x16;
typedef __attribute__((ext_vector_type(8)))  __bf16 bf16x8;
typedef __attribute__((ext_vector_type(8)))  float  f32x8;
typedef __attribute__((ext_vector_type(4)))  unsigned int u32x4;
typedef __attribute__((ext_vector_type(8)))  unsigned int u32x8;

static constexpr int B_BATCH = 256;
static constexpr int Z_DIM   = 256;
static constexpr int L_EXP   = 8;
static constexpr int O_DIM   = 8192;   // 128*8*8
static constexpr int K_MOE   = 2048;   // Z*L

__device__ __forceinline__ unsigned short f2bf(float f) {
  unsigned int u = __builtin_bit_cast(unsigned int, f);
  u += 0x7FFFu + ((u >> 16) & 1u);          // round-to-nearest-even
  return (unsigned short)(u >> 16);
}
__device__ __forceinline__ unsigned int pack2(unsigned short a, unsigned short b) {
  return (unsigned int)a | ((unsigned int)b << 16);
}
__device__ __forceinline__ bf16x16 make16(bf16x8 lo, bf16x8 hi) {
  return __builtin_shufflevector(lo, hi, 0,1,2,3,4,5,6,7,8,9,10,11,12,13,14,15);
}
// Low 32 bits of a generic pointer to LDS == LDS byte address (aperture rule).
__device__ __forceinline__ unsigned lds_addr_of(const void* p) {
  return (unsigned)(size_t)p;
}
// Async copy: one dword global -> LDS, tracked by ASYNCcnt.
__device__ __forceinline__ void async_g2l_b32(unsigned lds_dst, const void* gsrc) {
  asm volatile("global_load_async_to_lds_b32 %0, %1, off"
               :: "v"(lds_dst), "v"((unsigned long long)(size_t)gsrc)
               : "memory");
}
__device__ __forceinline__ void wait_asynccnt0() {
  asm volatile("s_wait_asynccnt 0x0" ::: "memory");
}
// TDM: 2D tile (tile_d0 x tile_d1 f32, row stride = stride_d0 elements) -> LDS.
__device__ __forceinline__ void tdm_load_tile_f32(unsigned lds_dst,
                                                  const void* gsrc,
                                                  unsigned tensor_d0, unsigned tensor_d1,
                                                  unsigned tile_d0, unsigned tile_d1,
                                                  unsigned stride_d0) {
  const unsigned long long ga = (unsigned long long)(size_t)gsrc;
  u32x4 g0;
  g0.x = 1u;                                                  // count=1 descriptor
  g0.y = lds_dst;                                             // lds_addr (bytes)
  g0.z = (unsigned)ga;                                        // global_addr[31:0]
  g0.w = (unsigned)((ga >> 32) & 0x1FFFFFFu) | (2u << 30);    // addr[56:32] | type=2
  u32x8 g1;
  g1[0] = 0x00020000u;                                        // data_size=4B
  g1[1] = (tensor_d0 & 0xFFFFu) << 16;                        // tensor_dim0[15:0]
  g1[2] = (tensor_d0 >> 16) | ((tensor_d1 & 0xFFFFu) << 16);  // dim0[31:16]|dim1[15:0]
  g1[3] = (tensor_d1 >> 16) | (tile_d0 << 16);                // dim1[31:16]|tile_dim0
  g1[4] = tile_d1;                                            // tile_dim1 (tile_dim2=0)
  g1[5] = stride_d0;                                          // tensor_dim0_stride[31:0]
  g1[6] = 0u;
  g1[7] = 0u;
  asm volatile("tensor_load_to_lds %0, %1" :: "s"(g0), "s"(g1) : "memory");
}

// ---------------------------------------------------------------------------
// Kernel 1: gating softmax + xg[b, i*8+l] = bf16(x[b,i] * g[b,l])
// ---------------------------------------------------------------------------
__global__ __launch_bounds__(256)
void gate_xg_kernel(const float* __restrict__ x, const float* __restrict__ gw,
                    const float* __restrict__ gb, float* __restrict__ g_out,
                    unsigned short* __restrict__ xg)
{
  __shared__ float red[256][L_EXP];
  __shared__ float logits[L_EXP];
  __shared__ float gsh[L_EXP];
  const int b = blockIdx.x, t = threadIdx.x;
  const float xv = x[b * Z_DIM + t];
#pragma unroll
  for (int l = 0; l < L_EXP; ++l) red[t][l] = xv * gw[t * L_EXP + l];
  __syncthreads();
  for (int s = 128; s > 0; s >>= 1) {
    if (t < s) {
#pragma unroll
      for (int l = 0; l < L_EXP; ++l) red[t][l] += red[t + s][l];
    }
    __syncthreads();
  }
  if (t < L_EXP) logits[t] = red[0][t] + gb[t];
  __syncthreads();
  if (t < L_EXP) {
    float m = logits[0];
#pragma unroll
    for (int i = 1; i < L_EXP; ++i) m = fmaxf(m, logits[i]);
    float s = 0.f;
#pragma unroll
    for (int i = 0; i < L_EXP; ++i) s += __expf(logits[i] - m);
    float gv = __expf(logits[t] - m) / s;
    gsh[t] = gv;
    g_out[b * L_EXP + t] = gv;
  }
  __syncthreads();
  unsigned short v[L_EXP];
#pragma unroll
  for (int l = 0; l < L_EXP; ++l) v[l] = f2bf(xv * gsh[l]);
  uint4 pv;
  pv.x = pack2(v[0], v[1]); pv.y = pack2(v[2], v[3]);
  pv.z = pack2(v[4], v[5]); pv.w = pack2(v[6], v[7]);
  *(uint4*)(xg + (size_t)b * K_MOE + t * L_EXP) = pv;
}

// ---------------------------------------------------------------------------
// Kernel 2: MoE GEMM  out[b,o] = sum_k xg[b,k]*pw[o,k]  + (g @ pb^T)[b,o]
// M=256 per block, N=16 per block, K=2048 in chunks of 128.
// B-tiles (16x128 f32) are double-buffered through the Tensor Data Mover:
// wave 0 keeps one TDM in flight (wait tensorcnt<=1) while the block converts
// and runs WMMA on the other buffer. A fragments load straight from global.
// ---------------------------------------------------------------------------
__global__ __launch_bounds__(256)
void moe_gemm_kernel(const unsigned short* __restrict__ xg,   // 256 x 2048 bf16
                     const float* __restrict__ pw,            // 8192 x 2048 f32
                     const float* __restrict__ pb,            // 8192 x 8 f32
                     const float* __restrict__ g,             // 256 x 8 f32
                     float* __restrict__ out)                 // 256 x 8192 f32
{
  constexpr int NCHUNK = K_MOE / 128;        // 16
  __shared__ float Braw[2][16][128];         // TDM destinations (ping-pong)
  __shared__ unsigned short Blds[2][16][136];// bf16 tiles (+pad)
  const int t = threadIdx.x;
  const int wave = t >> 5, lane = t & 31;
  const int ln = lane & 15, khalf = lane >> 4;
  const int n0 = blockIdx.x * 16;

  f32x8 acc0 = {0.f,0.f,0.f,0.f,0.f,0.f,0.f,0.f};
  f32x8 acc1 = {0.f,0.f,0.f,0.f,0.f,0.f,0.f,0.f};

  // prologue: TDM chunk 0 into buffer 0
  if (wave == 0) {
    tdm_load_tile_f32(lds_addr_of(&Braw[0][0][0]),
                      pw + (size_t)n0 * K_MOE,
                      K_MOE, O_DIM, 128, 16, K_MOE);
  }

  for (int ic = 0; ic < NCHUNK; ++ic) {
    const int cur = ic & 1;
    const int kc = ic * 128;
    if (wave == 0) {
      if (ic + 1 < NCHUNK) {
        // issue next chunk into the other buffer, then wait for current only
        tdm_load_tile_f32(lds_addr_of(&Braw[cur ^ 1][0][0]),
                          pw + (size_t)n0 * K_MOE + (kc + 128),
                          K_MOE, O_DIM, 128, 16, K_MOE);
        __builtin_amdgcn_s_wait_tensorcnt(1);
      } else {
        __builtin_amdgcn_s_wait_tensorcnt(0);
      }
    }
    __syncthreads();

    // ---- convert Braw[cur] (f32) -> Blds[cur] (bf16): 256 thr x 8 elems ----
    {
      const int n = t >> 4, q = t & 15;
      const float4 v0 = *(const float4*)&Braw[cur][n][q * 8];
      const float4 v1 = *(const float4*)&Braw[cur][n][q * 8 + 4];
      uint4 pv;
      pv.x = pack2(f2bf(v0.x), f2bf(v0.y));
      pv.y = pack2(f2bf(v0.z), f2bf(v0.w));
      pv.z = pack2(f2bf(v1.x), f2bf(v1.y));
      pv.w = pack2(f2bf(v1.z), f2bf(v1.w));
      *(uint4*)&Blds[cur][n][q * 8] = pv;
    }
    __syncthreads();

    // ---- 4 WMMA sub-steps of K=32 ----
#pragma unroll
    for (int s = 0; s < 4; ++s) {
      const int kb = kc + s * 32;
      bf16x8 blo = *(const bf16x8*)&Blds[cur][ln][s * 32 + khalf * 16];
      bf16x8 bhi = *(const bf16x8*)&Blds[cur][ln][s * 32 + khalf * 16 + 8];
      bf16x16 bfr = make16(blo, bhi);
      {
        const int row = wave * 16 + ln;
        const unsigned short* ap = xg + (size_t)row * K_MOE + kb + khalf * 16;
        bf16x8 alo = __builtin_bit_cast(bf16x8, *(const uint4*)(ap));
        bf16x8 ahi = __builtin_bit_cast(bf16x8, *(const uint4*)(ap + 8));
        acc0 = __builtin_amdgcn_wmma_f32_16x16x32_bf16(false, make16(alo, ahi),
                                                       false, bfr, (short)0,
                                                       acc0, false, false);
      }
      {
        const int row = (wave + 8) * 16 + ln;
        const unsigned short* ap = xg + (size_t)row * K_MOE + kb + khalf * 16;
        bf16x8 alo = __builtin_bit_cast(bf16x8, *(const uint4*)(ap));
        bf16x8 ahi = __builtin_bit_cast(bf16x8, *(const uint4*)(ap + 8));
        acc1 = __builtin_amdgcn_wmma_f32_16x16x32_bf16(false, make16(alo, ahi),
                                                       false, bfr, (short)0,
                                                       acc1, false, false);
      }
    }
    // The next iteration's first barrier separates these Blds reads from the
    // next convert pass; the DMA'd buffer was last read one barrier ago.
  }

  // ---- epilogue: += g[b,:] . pb[o,:] ----
  const int o = n0 + ln;
  float pbv[L_EXP];
  {
    const float4* pbp = (const float4*)(pb + (size_t)o * L_EXP);
    float4 p0 = pbp[0], p1 = pbp[1];
    pbv[0]=p0.x; pbv[1]=p0.y; pbv[2]=p0.z; pbv[3]=p0.w;
    pbv[4]=p1.x; pbv[5]=p1.y; pbv[6]=p1.z; pbv[7]=p1.w;
  }
#pragma unroll
  for (int r = 0; r < 8; ++r) {
    {
      const int b = wave * 16 + khalf * 8 + r;
      const float* gr = g + (size_t)b * L_EXP;
      float add = 0.f;
#pragma unroll
      for (int l = 0; l < L_EXP; ++l) add += gr[l] * pbv[l];
      out[(size_t)b * O_DIM + o] = acc0[r] + add;
    }
    {
      const int b = (wave + 8) * 16 + khalf * 8 + r;
      const float* gr = g + (size_t)b * L_EXP;
      float add = 0.f;
#pragma unroll
      for (int l = 0; l < L_EXP; ++l) add += gr[l] * pbv[l];
      out[(size_t)b * O_DIM + o] = acc1[r] + add;
    }
  }
}

// ---------------------------------------------------------------------------
// Kernel 3: ConvTranspose2d(k=4, s=2, p=1) as implicit GEMM per output pixel.
// Block = one output pixel: M=256 (batch), N=COUT, K=CIN*4 (zero-padded taps).
// Weights staged via async global->LDS copies (ASYNCcnt), then bf16-converted.
// ---------------------------------------------------------------------------
template<int CIN, int COUT, int HIN, bool RELU>
__global__ __launch_bounds__(256)
void convt_wmma_kernel(const float* __restrict__ in, const float* __restrict__ w,
                       const float* __restrict__ bias, float* __restrict__ out)
{
  constexpr int HOUT = 2 * HIN;
  constexpr int NT = (COUT + 15) / 16;
  constexpr int TILES = 16 * NT;
  constexpr int PER_WAVE = TILES / 8;
  constexpr int E = NT * 2;                 // async dwords per thread per chunk

  __shared__ unsigned short Alds[256][40];       // 256 x 32 bf16 (+pad)
  __shared__ float Wraw[NT * 16][32];            // raw f32 weights (async dest)
  __shared__ unsigned short Blds[NT * 16][40];   // bf16 weights (+pad)

  const int t = threadIdx.x;
  const int wave = t >> 5, lane = t & 31;
  const int ln = lane & 15, khalf = lane >> 4;

  const int p  = blockIdx.x;
  const int oy = p / HOUT, ox = p % HOUT;
  const int phy = (oy + 1) & 1, phx = (ox + 1) & 1;
  const int iy0 = (oy + 1 - phy) >> 1, ix0 = (ox + 1 - phx) >> 1;
  const bool vy0 = (iy0 >= 0) && (iy0 < HIN);
  const bool vy1 = (iy0 - 1 >= 0) && (iy0 - 1 < HIN);
  const bool vx0 = (ix0 >= 0) && (ix0 < HIN);
  const bool vx1 = (ix0 - 1 >= 0) && (ix0 - 1 < HIN);

  // this thread's weight-staging slice: E consecutive k's of one output chan
  const int widx0 = t * E;
  const int wn = widx0 >> 5;              // staged channel row (0..NT*16-1)
  const int wk0 = widx0 & 31;
  const int wosafe = (wn < COUT) ? wn : 0;

  f32x8 acc[PER_WAVE];
#pragma unroll
  for (int i = 0; i < PER_WAVE; ++i)
    acc[i] = (f32x8){0.f,0.f,0.f,0.f,0.f,0.f,0.f,0.f};

  for (int c0 = 0; c0 < CIN; c0 += 8) {
    // ---- stage A: thread t = batch image b; 8 channels x 4 taps -> bf16 ----
    {
      const float* ib = in + (size_t)t * CIN * HIN * HIN;
      unsigned short* arow = &Alds[t][0];
#pragma unroll
      for (int j = 0; j < 8; ++j) {
        const float* ic = ib + (size_t)(c0 + j) * HIN * HIN;
        if (j == 0) __builtin_prefetch(ic + 8 * HIN * HIN, 0, 0);
        float v00 = (vy0 && vx0) ? ic[iy0 * HIN + ix0]           : 0.f;
        float v01 = (vy0 && vx1) ? ic[iy0 * HIN + (ix0 - 1)]     : 0.f;
        float v10 = (vy1 && vx0) ? ic[(iy0 - 1) * HIN + ix0]     : 0.f;
        float v11 = (vy1 && vx1) ? ic[(iy0 - 1) * HIN + (ix0-1)] : 0.f;
        uint2 pv;
        pv.x = pack2(f2bf(v00), f2bf(v01));   // taps a=0,d=0..1
        pv.y = pack2(f2bf(v10), f2bf(v11));   // taps a=1,d=0..1
        *(uint2*)&arow[j * 4] = pv;
      }
    }
    // ---- stage W: async global -> LDS (raw f32), E dwords per thread ----
    {
#pragma unroll
      for (int e = 0; e < E; ++e) {
        const int k = wk0 + e;
        const int j = k >> 2, tap = k & 3;
        const int kh = phy + 2 * (tap >> 1), kw = phx + 2 * (tap & 1);
        const float* gp = w + (((size_t)(c0 + j) * COUT + wosafe) * 4 + kh) * 4 + kw;
        async_g2l_b32(lds_addr_of(&Wraw[wn][k]), gp);
      }
      wait_asynccnt0();
    }
    __syncthreads();

    // ---- convert Wraw -> Blds with out-of-range channel masking ----
    {
      const bool valid = (wn < COUT);
#pragma unroll
      for (int e = 0; e < E; e += 2) {
        const int k = wk0 + e;
        float a = valid ? Wraw[wn][k]     : 0.f;
        float b = valid ? Wraw[wn][k + 1] : 0.f;
        *(unsigned*)&Blds[wn][k] = pack2(f2bf(a), f2bf(b));
      }
    }
    __syncthreads();

    // ---- WMMA over this block's tile grid ----
#pragma unroll
    for (int i = 0; i < PER_WAVE; ++i) {
      const int tile = wave + i * 8;
      const int mt = tile & 15, nt = tile >> 4;
      const int arow = mt * 16 + ln;
      bf16x8 alo = *(const bf16x8*)&Alds[arow][khalf * 16];
      bf16x8 ahi = *(const bf16x8*)&Alds[arow][khalf * 16 + 8];
      bf16x16 af = make16(alo, ahi);
      const int brow = nt * 16 + ln;
      bf16x8 blo = *(const bf16x8*)&Blds[brow][khalf * 16];
      bf16x8 bhi = *(const bf16x8*)&Blds[brow][khalf * 16 + 8];
      bf16x16 bfr = make16(blo, bhi);
      acc[i] = __builtin_amdgcn_wmma_f32_16x16x32_bf16(false, af, false, bfr,
                                                       (short)0, acc[i], false, false);
    }
    __syncthreads();   // fragment reads must finish before next LDS overwrite
  }

  // ---- epilogue: bias (+ReLU), scatter to NCHW ----
#pragma unroll
  for (int i = 0; i < PER_WAVE; ++i) {
    const int tile = wave + i * 8;
    const int mt = tile & 15, nt = tile >> 4;
    const int o = nt * 16 + ln;
    if (o < COUT) {
      const float bv = bias[o];
#pragma unroll
      for (int r = 0; r < 8; ++r) {
        const int b = mt * 16 + khalf * 8 + r;
        float v = acc[i][r] + bv;
        if (RELU) v = fmaxf(v, 0.f);
        out[(((size_t)b * COUT + o) * HOUT + oy) * HOUT + ox] = v;
      }
    }
  }
}

// ---------------------------------------------------------------------------
// Host-side launch
// ---------------------------------------------------------------------------
extern "C" void kernel_launch(void* const* d_in, const int* in_sizes, int n_in,
                              void* d_out, int out_size, void* d_ws, size_t ws_size,
                              hipStream_t stream) {
  (void)in_sizes; (void)n_in; (void)out_size; (void)ws_size;
  const float* x  = (const float*)d_in[0];
  const float* gw = (const float*)d_in[1];
  const float* gb = (const float*)d_in[2];
  const float* pw = (const float*)d_in[3];
  const float* pb = (const float*)d_in[4];
  const float* w1 = (const float*)d_in[5];
  const float* b1 = (const float*)d_in[6];
  const float* w2 = (const float*)d_in[7];
  const float* b2 = (const float*)d_in[8];
  const float* w3 = (const float*)d_in[9];
  const float* b3 = (const float*)d_in[10];
  float* outp = (float*)d_out;

  // workspace layout
  char* ws = (char*)d_ws;
  size_t off = 0;
  float* g_buf = (float*)(ws + off);                off += (size_t)B_BATCH * L_EXP * 4;
  unsigned short* xg = (unsigned short*)(ws + off); off += (size_t)B_BATCH * K_MOE * 2;
  float* h0 = (float*)(ws + off);                   off += (size_t)B_BATCH * O_DIM * 4;
  float* h1 = (float*)(ws + off);                   off += (size_t)B_BATCH * 64 * 16 * 16 * 4;
  float* h2 = (float*)(ws + off);                   off += (size_t)B_BATCH * 32 * 32 * 32 * 4;

  // 1) gating + xg formation
  gate_xg_kernel<<<B_BATCH, 256, 0, stream>>>(x, gw, gb, g_buf, xg);

  // 2) MoE GEMM (M=256, N=8192, K=2048), double-buffered TDM B-tiles
  moe_gemm_kernel<<<O_DIM / 16, 256, 0, stream>>>(xg, pw, pb, g_buf, h0);

  // 3) conv-transpose decoder stack (implicit-GEMM WMMA, async weight staging)
  convt_wmma_kernel<128, 64, 8,  true ><<<16 * 16, 256, 0, stream>>>(h0, w1, b1, h1);
  convt_wmma_kernel<64,  32, 16, true ><<<32 * 32, 256, 0, stream>>>(h1, w2, b2, h2);
  convt_wmma_kernel<32,  3,  32, false><<<64 * 64, 256, 0, stream>>>(h2, w3, b3, outp);
}
